// BrainGraphEncoder_61529701483008
// MI455X (gfx1250) — matrix operand order repaired and checked by reading, not compiled, not run on hardware
//
#include <hip/hip_runtime.h>

typedef _Float16 v16h __attribute__((ext_vector_type(16)));
typedef _Float16 v8h  __attribute__((ext_vector_type(8)));
typedef float    v8f  __attribute__((ext_vector_type(8)));

#define N_REG   10
#define CH      7
#define HID     128
#define HEADS   4
#define NL      3
#define TILE_TOK 8
#define ROWS    (TILE_TOK * N_REG)   // 80 graph rows per workgroup
#define TOK     16384                // B*T = 16*1024
#define NBLK    (TOK / TILE_TOK)     // 2048 workgroups
#define NPAD    136                  // padded f16 row stride for node features
#define HHPAD   520                  // padded f16 row stride for hh (512 + 8)
#define NWIDE   (HEADS * HID)        // 512

union H16 { v16h v; v8h h[2]; };

__device__ __forceinline__ float gelu_exact(float x) {
    return 0.5f * x * (1.0f + erff(x * 0.70710678118654752440f));
}

// ---------------------------------------------------------------------------
// Prep: Wt[l][n][k] = (f16) W_gat[l][k][n],  n = h*128+e, so WMMA B tiles are
// contiguous 32B per lane per K-step.
// ---------------------------------------------------------------------------
__global__ void prep_wt_kernel(const float* __restrict__ Wg,
                               _Float16* __restrict__ Wt) {
    int idx = blockIdx.x * 256 + threadIdx.x;       // 3*512*128 = 196608
    if (idx >= NL * NWIDE * HID) return;
    int l = idx / (NWIDE * HID);
    int rem = idx - l * (NWIDE * HID);
    int n = rem / HID;
    int k = rem - n * HID;
    Wt[idx] = (_Float16)Wg[l * (HID * NWIDE) + k * NWIDE + n];
}

// ---------------------------------------------------------------------------
// Fused encoder + 3x GAT. One workgroup (8 waves) = 8 tokens (80 nodes).
// ---------------------------------------------------------------------------
__global__ void __launch_bounds__(256)
brain_graph_kernel(const float* __restrict__ x,
                   const float* __restrict__ W_enc,
                   const float* __restrict__ b_enc,
                   const float* __restrict__ g_enc,
                   const float* __restrict__ beta_enc,
                   const _Float16* __restrict__ Wt,
                   const float* __restrict__ att_src,
                   const float* __restrict__ att_dst,
                   const float* __restrict__ b_gat,
                   float* __restrict__ out_gf,
                   float* __restrict__ out_enc) {
    // adjacency row masks: bit j of ADJ[i] => edge j -> i (incl. self loop)
    const unsigned ADJ[N_REG] = {0x01Fu, 0x0A7u, 0x08Fu, 0x00Du, 0x031u,
                                 0x072u, 0x260u, 0x186u, 0x180u, 0x240u};

    __shared__ __attribute__((aligned(16))) _Float16 s_nodes_h[ROWS][NPAD];
    __shared__ float    s_nodes_f[ROWS][HID];
    __shared__ __attribute__((aligned(16))) _Float16 s_hh[ROWS][HHPAD];
    __shared__ float    s_as[ROWS][HEADS];
    __shared__ float    s_ad[ROWS][HEADS];
    __shared__ float    s_alpha[TILE_TOK][HEADS][N_REG][N_REG];

    const int tid  = threadIdx.x;
    const int wave = tid >> 5;
    const int lane = tid & 31;
    const int r16  = lane & 15;
    const int half = lane >> 4;

    // ------------------- region encoders: Linear(7,128)+LN+GELU ------------
    // one wave per (token, region) pair; lane covers 4 of 128 dims
    for (int pair = wave; pair < ROWS; pair += 8) {
        int token_l = pair / N_REG;
        int r = pair - token_l * N_REG;
        size_t tg = (size_t)blockIdx.x * TILE_TOK + token_l;
        int d0 = lane * 4;
        float hv[4];
        #pragma unroll
        for (int dd = 0; dd < 4; ++dd) hv[dd] = b_enc[r * HID + d0 + dd];
        #pragma unroll
        for (int c = 0; c < CH; ++c) {
            float xv = x[tg * (N_REG * CH) + r * CH + c];
            const float* wr = W_enc + (size_t)(r * CH + c) * HID + d0;
            #pragma unroll
            for (int dd = 0; dd < 4; ++dd) hv[dd] = fmaf(xv, wr[dd], hv[dd]);
        }
        float s = 0.f, ss = 0.f;
        #pragma unroll
        for (int dd = 0; dd < 4; ++dd) { s += hv[dd]; ss += hv[dd] * hv[dd]; }
        #pragma unroll
        for (int off = 16; off >= 1; off >>= 1) {
            s  += __shfl_xor(s, off, 32);
            ss += __shfl_xor(ss, off, 32);
        }
        float mu  = s * (1.0f / 128.0f);
        float var = ss * (1.0f / 128.0f) - mu * mu;
        float rstd = rsqrtf(var + 1e-5f);
        #pragma unroll
        for (int dd = 0; dd < 4; ++dd) {
            int d = d0 + dd;
            float t = (hv[dd] - mu) * rstd * g_enc[r * HID + d] + beta_enc[r * HID + d];
            float g = gelu_exact(t);
            s_nodes_f[pair][d] = g;
            s_nodes_h[pair][d] = (_Float16)g;
            out_enc[tg * (N_REG * HID) + r * HID + d] = g;
        }
    }
    __syncthreads();

    // ------------------------------- GAT layers ----------------------------
    for (int l = 0; l < NL; ++l) {
        // ---- hh = nodes(80x128) @ W(128x512) via v_wmma_f32_16x16x32_f16 ---
        {
            const _Float16* Wl = Wt + (size_t)l * (NWIDE * HID);
            #pragma unroll
            for (int q = 0; q < 4; ++q) {
                int n0 = (wave + 8 * q) * 16;
                H16 Bv[4];
                const _Float16* bp = Wl + (size_t)(n0 + r16) * HID + half * 16;
                #pragma unroll
                for (int kk = 0; kk < 4; ++kk)
                    Bv[kk].v = *(const v16h*)(bp + kk * 32);
                #pragma unroll
                for (int m = 0; m < 5; ++m) {
                    int m0 = m * 16;
                    const _Float16* ap = &s_nodes_h[m0 + r16][0];
                    H16 Av[4];
                    #pragma unroll
                    for (int kk = 0; kk < 4; ++kk) {
                        Av[kk].h[0] = *(const v8h*)(ap + kk * 32 + half * 16);
                        Av[kk].h[1] = *(const v8h*)(ap + kk * 32 + half * 16 + 8);
                    }
                    v8f acc = {};
                    #pragma unroll
                    for (int kk = 0; kk < 4; ++kk)
                        acc = __builtin_amdgcn_wmma_f32_16x16x32_f16(
                            false, Av[kk].v, false, Bv[kk].v,
                            (short)0, acc, false, false);
                    #pragma unroll
                    for (int v = 0; v < 8; ++v)
                        s_hh[m0 + v + 8 * half][n0 + r16] = (_Float16)acc[v];
                }
            }
        }
        __syncthreads();

        // ---- attention logit dot products: a_s, a_d ------------------------
        for (int p = tid; p < ROWS * HEADS; p += 256) {
            int n = p >> 2, h = p & 3;
            const float* ws = att_src + (size_t)(l * HEADS + h) * HID;
            const float* wd = att_dst + (size_t)(l * HEADS + h) * HID;
            const _Float16* hrow = &s_hh[n][h * HID];
            float as = 0.f, ad = 0.f;
            #pragma unroll 8
            for (int e = 0; e < HID; ++e) {
                float hv = (float)hrow[e];
                as = fmaf(hv, ws[e], as);
                ad = fmaf(hv, wd[e], ad);
            }
            s_as[n][h] = as;
            s_ad[n][h] = ad;
        }
        __syncthreads();

        // ---- masked leaky-relu softmax over 10 sources ---------------------
        for (int p = tid; p < TILE_TOK * HEADS * N_REG; p += 256) {
            int token = p / (HEADS * N_REG);
            int rem = p - token * (HEADS * N_REG);
            int i = rem >> 2, h = rem & 3;
            float ad = s_ad[token * N_REG + i][h];
            float lg[N_REG];
            float mx = -3.0e38f;
            #pragma unroll
            for (int j = 0; j < N_REG; ++j) {
                float v = ad + s_as[token * N_REG + j][h];
                v = (v >= 0.f) ? v : 0.2f * v;                 // leaky relu
                if (!((ADJ[i] >> j) & 1u)) v = -1e9f;           // mask
                lg[j] = v;
                mx = fmaxf(mx, v);
            }
            float sum = 0.f;
            #pragma unroll
            for (int j = 0; j < N_REG; ++j) { lg[j] = expf(lg[j] - mx); sum += lg[j]; }
            float inv = 1.0f / sum;
            #pragma unroll
            for (int j = 0; j < N_REG; ++j)
                s_alpha[token][h][i][j] = lg[j] * inv;
        }
        __syncthreads();

        // ---- aggregate, head-mean, bias, GELU, residual --------------------
        for (int idx = tid; idx < ROWS * HID; idx += 256) {
            int row = idx >> 7;
            int e = idx & 127;
            int token = row / N_REG;
            int i = row - token * N_REG;
            float acc = 0.f;
            #pragma unroll
            for (int h = 0; h < HEADS; ++h) {
                #pragma unroll
                for (int j = 0; j < N_REG; ++j)
                    acc = fmaf(s_alpha[token][h][i][j],
                               (float)s_hh[token * N_REG + j][h * HID + e], acc);
            }
            acc = acc * 0.25f + b_gat[l * HID + e];
            float nv = gelu_exact(acc) + s_nodes_f[row][e];
            s_nodes_f[row][e] = nv;
            s_nodes_h[row][e] = (_Float16)nv;
            if (l == NL - 1) {
                size_t tg = (size_t)blockIdx.x * TILE_TOK + token;
                out_gf[tg * (N_REG * HID) + i * HID + e] = nv;
            }
        }
        __syncthreads();
    }
}

extern "C" void kernel_launch(void* const* d_in, const int* in_sizes, int n_in,
                              void* d_out, int out_size, void* d_ws, size_t ws_size,
                              hipStream_t stream) {
    const float* x        = (const float*)d_in[0];
    const float* W_enc    = (const float*)d_in[1];
    const float* b_enc    = (const float*)d_in[2];
    const float* g_enc    = (const float*)d_in[3];
    const float* beta_enc = (const float*)d_in[4];
    const float* W_gat    = (const float*)d_in[5];
    const float* att_src  = (const float*)d_in[6];
    const float* att_dst  = (const float*)d_in[7];
    const float* b_gat    = (const float*)d_in[8];

    _Float16* Wt = (_Float16*)d_ws;                 // 3*512*128 f16 = 384 KB

    float* out_gf  = (float*)d_out;
    float* out_enc = (float*)d_out + (size_t)TOK * (N_REG * HID);

    prep_wt_kernel<<<(NL * NWIDE * HID + 255) / 256, 256, 0, stream>>>(W_gat, Wt);
    brain_graph_kernel<<<NBLK, 256, 0, stream>>>(
        x, W_enc, b_enc, g_enc, beta_enc, Wt, att_src, att_dst, b_gat,
        out_gf, out_enc);
}
